// Decoder_78486232367391
// MI455X (gfx1250) — compile-verified, matrix operand order checked by reading, not run on hardware
//
#include <hip/hip_runtime.h>
#include <hip/hip_bf16.h>

// ---------------------------------------------------------------------------
// Types for CDNA5 WMMA (wave32): bf16 A/B fragments, f32 C/D accumulator.
// ---------------------------------------------------------------------------
typedef __bf16 v16bf __attribute__((ext_vector_type(16)));
typedef __bf16 v8bf  __attribute__((ext_vector_type(8)));
typedef float  v8f   __attribute__((ext_vector_type(8)));
typedef int    i32x4 __attribute__((vector_size(16)));

#define WMMA_BF16(a, b, c) \
    __builtin_amdgcn_wmma_f32_16x16x32_bf16(false, (a), false, (b), (short)0, (c), false, false)

// CDNA5 async global->LDS path (ASYNCcnt-tracked), guarded so the file still
// compiles on toolchains that don't declare the builtins.
#if defined(__HIP_DEVICE_COMPILE__) && \
    __has_builtin(__builtin_amdgcn_global_load_async_to_lds_b128)
#define HAVE_ASYNC_LDS 1
#else
#define HAVE_ASYNC_LDS 0
#endif

// Problem constants
#define BB 48
#define EE 512
#define HH 1024
#define VV 32000
#define TT 64

// float -> bf16 (round to nearest even)
__device__ __forceinline__ unsigned short f2bf(float f) {
    unsigned int u = __float_as_uint(f);
    unsigned int r = u + 0x7FFFu + ((u >> 16) & 1u);
    return (unsigned short)(r >> 16);
}

__device__ __forceinline__ float sigmoidf_(float x) { return 1.0f / (1.0f + expf(-x)); }

// Load one 16x32 bf16 A/B fragment per ISA layout:
// lane L holds row (L&15); the 16 bf16 elements are the two contiguous
// 8-element chunks at k = kb + 8*(L>>4) and k = kb + 8*(L>>4) + 16.
// `row` points at the start of this lane's row (row-major, K contiguous).
__device__ __forceinline__ v16bf load_frag(const unsigned short* __restrict__ row,
                                           int kb, int h) {
    const unsigned short* p = row + kb + h * 8;
    v8bf lo = *(const v8bf*)(p);
    v8bf hi = *(const v8bf*)(p + 16);
    return __builtin_shufflevector(lo, hi, 0, 1, 2, 3, 4, 5, 6, 7,
                                   8, 9, 10, 11, 12, 13, 14, 15);
}

// Same fragment gather, but from an LDS-staged 32-wide K-slice (kb == 0).
__device__ __forceinline__ v16bf lds_frag(const unsigned short* row, int h) {
    const unsigned short* p = row + h * 8;
    v8bf lo = *(const v8bf*)(p);
    v8bf hi = *(const v8bf*)(p + 16);
    return __builtin_shufflevector(lo, hi, 0, 1, 2, 3, 4, 5, 6, 7,
                                   8, 9, 10, 11, 12, 13, 14, 15);
}

__device__ __forceinline__ void stage_wait_async() {
#if HAVE_ASYNC_LDS
#if __has_builtin(__builtin_amdgcn_s_wait_asynccnt)
    __builtin_amdgcn_s_wait_asynccnt(0);
#else
    asm volatile("s_wait_asynccnt 0x0" ::: "memory");
#endif
#endif
}

// ---------------------------------------------------------------------------
// f32 -> bf16 weight conversion (n must be a multiple of 4)
// ---------------------------------------------------------------------------
__global__ void k_cvt_bf16(const float* __restrict__ src,
                           unsigned short* __restrict__ dst, int n) {
    int i = (blockIdx.x * blockDim.x + threadIdx.x) * 4;
    if (i < n) {
        float4 f = *(const float4*)(src + i);
        dst[i + 0] = f2bf(f.x);
        dst[i + 1] = f2bf(f.y);
        dst[i + 2] = f2bf(f.z);
        dst[i + 3] = f2bf(f.w);
    }
}

// ---------------------------------------------------------------------------
// Init: h0 = latent @ W_lh^T + b_lh ; peep = latent @ Wp^T + bp (constant over
// time, hoisted like the reference); emb[0] = start ; stops[:,0] = 0.
// ---------------------------------------------------------------------------
__global__ void k_init(const float* __restrict__ latent,
                       const float* __restrict__ W_lh, const float* __restrict__ b_lh,
                       const float* __restrict__ Wp, const float* __restrict__ bp,
                       const float* __restrict__ start,
                       float* __restrict__ hF0, unsigned short* __restrict__ hH0,
                       float* __restrict__ peep,
                       unsigned short* __restrict__ embh0,
                       float* __restrict__ stops) {
    int idx = blockIdx.x * blockDim.x + threadIdx.x;
    int b = idx >> 12;            // / 4096
    int j = idx & 4095;
    if (b >= BB) return;
    const float* lrow = latent + b * HH;
    if (j < HH) {
        const float* wrow = W_lh + j * HH;
        float acc = 0.f;
#pragma unroll 4
        for (int k = 0; k < HH; ++k) acc += lrow[k] * wrow[k];
        float hv = acc + b_lh[j];
        hF0[b * HH + j] = hv;
        hH0[b * HH + j] = f2bf(hv);
    } else {
        int jj = j - HH;          // 0..3071
        const float* wrow = Wp + jj * HH;
        float acc = 0.f;
#pragma unroll 4
        for (int k = 0; k < HH; ++k) acc += lrow[k] * wrow[k];
        peep[b * 3 * HH + jj] = acc + bp[jj];
    }
    if (j < EE) embh0[b * EE + j] = f2bf(start[j]);
    if (j == 0) stops[b * TT] = 0.f;     // stops[b][0]
}

// ---------------------------------------------------------------------------
// GRU gates: hy = GRU(emb_prev, h_prev) with peephole.
// One wave per 16x16 tile of hy: 3 M-tiles x 64 N-tiles = 192 waves.
// r & i gates combine ig+hg in one accumulator; n gate needs ig and hg
// separate (resetgate multiplies only the hg part).
// ---------------------------------------------------------------------------
__global__ void __launch_bounds__(128)
k_gates(const unsigned short* __restrict__ embPrev,   // (48, 512) bf16
        const float* __restrict__ hFprev,             // (48, 1024) f32
        const unsigned short* __restrict__ hHprev,    // (48, 1024) bf16
        const unsigned short* __restrict__ WiH,       // (3072, 512) bf16
        const unsigned short* __restrict__ WhH,       // (3072, 1024) bf16
        const float* __restrict__ bi, const float* __restrict__ bh,
        const float* __restrict__ peep,               // (48, 3072) f32
        float* __restrict__ hFnext,                   // (48, 1024) f32
        unsigned short* __restrict__ hHnext) {        // (48, 1024) bf16
    int lane = threadIdx.x & 31, w = threadIdx.x >> 5;
    int g  = blockIdx.x * 4 + w;          // 0..191
    int mi = g % 3, nj = g / 3;           // M-tile, N-tile
    int h = lane >> 4, r = lane & 15;

    const unsigned short* aE = embPrev + (mi * 16 + r) * EE;
    const unsigned short* aH = hHprev + (mi * 16 + r) * HH;
    int n = nj * 16 + r;                  // weight row (= output column)
    const unsigned short* wiR = WiH + (size_t)(n) * EE;
    const unsigned short* wiI = WiH + (size_t)(HH + n) * EE;
    const unsigned short* wiN = WiH + (size_t)(2 * HH + n) * EE;
    const unsigned short* whR = WhH + (size_t)(n) * HH;
    const unsigned short* whI = WhH + (size_t)(HH + n) * HH;
    const unsigned short* whN = WhH + (size_t)(2 * HH + n) * HH;

    v8f cr = {0, 0, 0, 0, 0, 0, 0, 0}, ci = cr, cin = cr, chn = cr;

    for (int k = 0; k < EE; k += 32) {          // ig = emb @ Wi^T
        v16bf a = load_frag(aE, k, h);
        cr  = WMMA_BF16(a, load_frag(wiR, k, h), cr);
        ci  = WMMA_BF16(a, load_frag(wiI, k, h), ci);
        cin = WMMA_BF16(a, load_frag(wiN, k, h), cin);
    }
    for (int k = 0; k < HH; k += 32) {          // hg = h @ Wh^T
        v16bf a = load_frag(aH, k, h);
        cr  = WMMA_BF16(a, load_frag(whR, k, h), cr);
        ci  = WMMA_BF16(a, load_frag(whI, k, h), ci);
        chn = WMMA_BF16(a, load_frag(whN, k, h), chn);
    }

    int jg = nj * 16 + r;                       // C/D layout: n = lane&15
#pragma unroll
    for (int e = 0; e < 8; ++e) {
        int b = mi * 16 + e + 8 * h;            // C/D layout: m = e + 8*(lane>>4)
        const float* prow = peep + b * 3 * HH;
        float xr = cr[e] + bi[jg] + bh[jg] + prow[jg];
        float xi = ci[e] + bi[HH + jg] + bh[HH + jg] + prow[HH + jg];
        float hn = chn[e] + bh[2 * HH + jg];
        float rg = sigmoidf_(xr);
        float ig = sigmoidf_(xi);
        float ng = tanhf(cin[e] + bi[2 * HH + jg] + rg * hn + prow[2 * HH + jg]);
        float hp = hFprev[b * HH + jg];
        float hy = ng + ig * (hp - ng);
        hFnext[b * HH + jg] = hy;
        hHnext[b * HH + jg] = f2bf(hy);
    }
}

// ---------------------------------------------------------------------------
// Output projection: o = hy @ Wo^T + bo.
// Blocks 0..23 (96 waves = 3 M-tiles x 32 N-tiles) do the 512 emb columns
// with WMMA -> emb[t] = tanh(o) stored as bf16.
// Block 24 does the single stop column in f32 (48 dots of K=1024).
// ---------------------------------------------------------------------------
__global__ void __launch_bounds__(128)
k_out(const unsigned short* __restrict__ hyH,     // (48, 1024) bf16
      const float* __restrict__ hyF,              // (48, 1024) f32
      const unsigned short* __restrict__ WoH,     // (512, 1024) bf16
      const float* __restrict__ Wo,               // (513, 1024) f32
      const float* __restrict__ bo,               // (513)
      unsigned short* __restrict__ embOut,        // (48, 512) bf16, slice t
      float* __restrict__ stops, int t) {
    if (blockIdx.x == 24) {                       // stop column (uniform branch)
        int tid = threadIdx.x;
        if (tid < BB) {
            const float* wrow = Wo + (size_t)EE * HH;   // row 512
            const float* hrow = hyF + tid * HH;
            float acc = 0.f;
#pragma unroll 4
            for (int k = 0; k < HH; ++k) acc += hrow[k] * wrow[k];
            stops[tid * TT + t] = sigmoidf_(acc + bo[EE]);
        }
        return;
    }
    int lane = threadIdx.x & 31, w = threadIdx.x >> 5;
    int g  = blockIdx.x * 4 + w;                  // 0..95
    int mi = g % 3, nj = g / 3;                   // nj: 0..31
    int h = lane >> 4, r = lane & 15;

    const unsigned short* aH = hyH + (mi * 16 + r) * HH;
    const unsigned short* bW = WoH + (size_t)(nj * 16 + r) * HH;
    v8f c = {0, 0, 0, 0, 0, 0, 0, 0};
    for (int k = 0; k < HH; k += 32) {
        v16bf a = load_frag(aH, k, h);
        c = WMMA_BF16(a, load_frag(bW, k, h), c);
    }
    int jg = nj * 16 + r;
#pragma unroll
    for (int e = 0; e < 8; ++e) {
        int b = mi * 16 + e + 8 * h;
        float em = tanhf(c[e] + bo[jg]);
        embOut[b * EE + jg] = f2bf(em);
    }
}

// ---------------------------------------------------------------------------
// Unembed for all 64 timesteps at once: logits = emb @ Wu^T + bu.
// Rows r = t*48 + b (3072 rows; 16-row tiles never cross a t boundary).
// Block = 4 waves; each wave: 2 M-tiles x 4 N-tiles (8 WMMA per K-step).
// The 64 Wu rows of the block's N-group are staged into LDS (80B-padded rows,
// conflict-free ds_load_b128), double-buffered, with CDNA5 async
// global->LDS copies overlapping the WMMAs of the previous K-slice.
// Grid: x = 500 N-groups (64 cols), y = 24 (8 M-tiles per block).
// ---------------------------------------------------------------------------
#define BSTR 40   // ushorts per staged row: 32 data + 8 pad (80 B)

__device__ __forceinline__ void stage_slice(unsigned short* bufBase,
                                            const unsigned short* __restrict__ wuRow0,
                                            int kb, int tid) {
#pragma unroll
    for (int i = 0; i < 2; ++i) {
        int c = tid + i * 128;               // 256 16B-chunks: 64 rows x 4
        int row = c >> 2, sub = c & 3;
        const unsigned short* g = wuRow0 + (size_t)row * EE + kb + sub * 8;
        unsigned short* l = bufBase + row * BSTR + sub * 8;
#if HAVE_ASYNC_LDS
        __builtin_amdgcn_global_load_async_to_lds_b128(
            (__attribute__((address_space(1))) i32x4*)(g),
            (__attribute__((address_space(3))) i32x4*)(l), 0, 0);
#else
        *(v8bf*)l = *(const v8bf*)g;
#endif
    }
}

__global__ void __launch_bounds__(128)
k_unembed(const unsigned short* __restrict__ embh,   // (3072, 512) bf16
          const unsigned short* __restrict__ WuH,    // (32000, 512) bf16
          const float* __restrict__ bu,              // (32000)
          float* __restrict__ out) {                 // (48, 64, 32000) f32
    __shared__ unsigned short Bs[2][64 * BSTR];      // 2 x 5 KB

    int tid = threadIdx.x;
    int lane = tid & 31, w = tid >> 5;
    int h = lane >> 4, r = lane & 15;
    int mg = (blockIdx.y * 4 + w) * 2;               // first of 2 M-tiles (0..190)
    int n0 = blockIdx.x * 64;

    const unsigned short* a0 = embh + (size_t)(mg * 16 + r) * EE;
    const unsigned short* a1 = a0 + (size_t)16 * EE;
    const unsigned short* wuRow0 = WuH + (size_t)n0 * EE;

    v8f z = {0, 0, 0, 0, 0, 0, 0, 0};
    v8f acc0[4] = {z, z, z, z};
    v8f acc1[4] = {z, z, z, z};

    stage_slice(&Bs[0][0], wuRow0, 0, tid);          // prologue: slice 0
    for (int ks = 0; ks < 16; ++ks) {
        int cur = ks & 1;
        stage_wait_async();                          // own slice-ks copies done
        __syncthreads();                             // all waves' copies visible;
                                                     // also fences last consume
        if (ks + 1 < 16)
            stage_slice(&Bs[1 - cur][0], wuRow0, (ks + 1) * 32, tid);

        v16bf aA = load_frag(a0, ks * 32, h);
        v16bf aB = load_frag(a1, ks * 32, h);
#pragma unroll
        for (int g = 0; g < 4; ++g) {
            v16bf bf = lds_frag(&Bs[cur][(16 * g + r) * BSTR], h);
            acc0[g] = WMMA_BF16(aA, bf, acc0[g]);
            acc1[g] = WMMA_BF16(aB, bf, acc1[g]);
        }
    }

#pragma unroll
    for (int m = 0; m < 2; ++m) {
        int mi = mg + m;
        int t = mi / 3, bbase = (mi % 3) * 16;
        const v8f* acc = m ? acc1 : acc0;
#pragma unroll
        for (int e = 0; e < 8; ++e) {
            int b = bbase + e + 8 * h;
            float* orow = out + (size_t)b * TT * VV + (size_t)t * VV;
            int v0 = n0 + r;
            orow[v0 + 0]  = acc[0][e] + bu[v0 + 0];
            orow[v0 + 16] = acc[1][e] + bu[v0 + 16];
            orow[v0 + 32] = acc[2][e] + bu[v0 + 32];
            orow[v0 + 48] = acc[3][e] + bu[v0 + 48];
        }
    }
}

// ---------------------------------------------------------------------------
// Host side
// ---------------------------------------------------------------------------
extern "C" void kernel_launch(void* const* d_in, const int* in_sizes, int n_in,
                              void* d_out, int out_size, void* d_ws, size_t ws_size,
                              hipStream_t stream) {
    (void)in_sizes; (void)n_in; (void)out_size; (void)ws_size;
    const float* latent = (const float*)d_in[1];
    const float* Wi   = (const float*)d_in[2];
    const float* bi   = (const float*)d_in[3];
    const float* Wh   = (const float*)d_in[4];
    const float* bh   = (const float*)d_in[5];
    const float* Wp   = (const float*)d_in[6];
    const float* bp   = (const float*)d_in[7];
    const float* W_lh = (const float*)d_in[8];
    const float* b_lh = (const float*)d_in[9];
    const float* Wo   = (const float*)d_in[10];
    const float* bo   = (const float*)d_in[11];
    const float* Wu   = (const float*)d_in[12];
    const float* bu   = (const float*)d_in[13];
    const float* start = (const float*)d_in[14];

    float* out   = (float*)d_out;
    float* stops = out + (size_t)BB * TT * VV;

    // Workspace carve-up (256B aligned)
    char* ws = (char*)d_ws;
    size_t off = 0;
    auto carve = [&](size_t bytes) -> void* {
        void* p = ws + off;
        off = (off + bytes + 255) & ~(size_t)255;
        return p;
    };
    unsigned short* WuH  = (unsigned short*)carve((size_t)VV * EE * 2);      // 32 MB
    unsigned short* WiH  = (unsigned short*)carve((size_t)3 * HH * EE * 2);  // 3 MB
    unsigned short* WhH  = (unsigned short*)carve((size_t)3 * HH * HH * 2);  // 6 MB
    unsigned short* WoH  = (unsigned short*)carve((size_t)EE * HH * 2);      // 1 MB
    unsigned short* embh = (unsigned short*)carve((size_t)TT * BB * EE * 2); // 3 MB
    unsigned short* hH0  = (unsigned short*)carve((size_t)BB * HH * 2);
    unsigned short* hH1  = (unsigned short*)carve((size_t)BB * HH * 2);
    float* hF0  = (float*)carve((size_t)BB * HH * 4);
    float* hF1  = (float*)carve((size_t)BB * HH * 4);
    float* peep = (float*)carve((size_t)BB * 3 * HH * 4);

    unsigned short* hH[2] = {hH0, hH1};
    float*          hF[2] = {hF0, hF1};

    // 1) weight conversion to bf16 (all counts are multiples of 4)
    {
        int nWu = VV * EE, nWi = 3 * HH * EE, nWh = 3 * HH * HH, nWo = EE * HH;
        k_cvt_bf16<<<(nWu / 4 + 255) / 256, 256, 0, stream>>>(Wu, WuH, nWu);
        k_cvt_bf16<<<(nWi / 4 + 255) / 256, 256, 0, stream>>>(Wi, WiH, nWi);
        k_cvt_bf16<<<(nWh / 4 + 255) / 256, 256, 0, stream>>>(Wh, WhH, nWh);
        k_cvt_bf16<<<(nWo / 4 + 255) / 256, 256, 0, stream>>>(Wo, WoH, nWo);
    }

    // 2) init: h0, peep, emb[0], stops[:,0]
    k_init<<<(BB * 4096) / 256, 256, 0, stream>>>(latent, W_lh, b_lh, Wp, bp, start,
                                                  hF[0], hH[0], peep, embh, stops);

    // 3) recurrence
    for (int t = 1; t < TT; ++t) {
        int pp = (t - 1) & 1, cc = t & 1;
        k_gates<<<48, 128, 0, stream>>>(embh + (size_t)(t - 1) * BB * EE,
                                        hF[pp], hH[pp], WiH, WhH, bi, bh, peep,
                                        hF[cc], hH[cc]);
        k_out<<<25, 128, 0, stream>>>(hH[cc], hF[cc], WoH, Wo, bo,
                                      embh + (size_t)t * BB * EE, stops, t);
    }

    // 4) unembed all timesteps: (3072 x 32000, K=512), LDS-staged B
    k_unembed<<<dim3(500, 24), 128, 0, stream>>>(embh, WuH, bu, out);
}
